// Encoder_45286135169025
// MI455X (gfx1250) — compile-verified, matrix operand order checked by reading, not run on hardware
//
#include <hip/hip_runtime.h>
#include <math.h>

// ---------------- types ----------------
typedef __attribute__((ext_vector_type(16))) __bf16 v16bf;
typedef __attribute__((ext_vector_type(8)))  __bf16 v8bf;
typedef __attribute__((ext_vector_type(8)))  float  v8f;

#define BDIM 8
#define TDIM 1024
#define CDIM 192
#define LNUM 6
#define KCD  96
#define FCD  768
#define MROWS (BDIM*TDIM)   // 8192

__device__ __forceinline__ v8f zero8() {
  v8f r;
#pragma unroll
  for (int i = 0; i < 8; ++i) r[i] = 0.f;
  return r;
}
__device__ __forceinline__ v16bf zero16() {
  v16bf r;
#pragma unroll
  for (int i = 0; i < 16; ++i) r[i] = (__bf16)0.f;
  return r;
}
__device__ __forceinline__ v16bf mk16(v8bf lo, v8bf hi) {
  v16bf r;
#pragma unroll
  for (int i = 0; i < 8; ++i) { r[i] = lo[i]; r[i + 8] = hi[i]; }
  return r;
}
// loads K 0..7 (+half8) and K 16..23 (+half8): contiguous 16B chunks
__device__ __forceinline__ v16bf ldtile(const __bf16* p) {
  return mk16(*(const v8bf*)p, *(const v8bf*)(p + 16));
}
__device__ __forceinline__ v8f wmma_bf(v16bf a, v16bf b, v8f c) {
  return __builtin_amdgcn_wmma_f32_16x16x32_bf16(false, a, false, b, (short)0, c,
                                                 false, false);
}

// ---------------- generic GEMM (bf16 WMMA, f32 acc), compile-time shapes -------
// out[m, n] = (sum_k A'[m,k] * W[n,k] + bias[n]) * scale ; optional relu.
// A' row m=(b*T+t): concat over TAPS of A[b*T + t+tap-TAPS/2, 0..CIN) (zeros OOB).
// grid = (M/128, NOUT/64), block = 256 (8 waves).
// Each wave owns a 16x64 strip: one A operand feeds 4 WMMAs per K-chunk.
template <int CIN, int TAPS, int NOUT>
__global__ __launch_bounds__(256)
void gemm_tpl(const __bf16* __restrict__ A, const __bf16* __restrict__ W,
              const float* __restrict__ bias, float* __restrict__ outF,
              __bf16* __restrict__ outB, float scale, int relu) {
  constexpr int KTOT = TAPS * CIN;
  const int lane  = threadIdx.x & 31;
  const int wave  = threadIdx.x >> 5;
  const int mbase = blockIdx.x * 128 + wave * 16;
  const int nbase = blockIdx.y * 64;
  const int half8 = (lane & 16) ? 8 : 0;
  const int lrow  = lane & 15;
  const int grow  = mbase + lrow;
  const int t     = grow & (TDIM - 1);      // T = 1024, batches contiguous
  const __bf16* arow = A + (size_t)grow * CIN + half8;
  const __bf16* wrow[4];
#pragma unroll
  for (int i = 0; i < 4; ++i)
    wrow[i] = W + (size_t)(nbase + i * 16 + lrow) * KTOT + half8;
  v8f acc[4];
#pragma unroll
  for (int i = 0; i < 4; ++i) acc[i] = zero8();

#pragma unroll
  for (int tap = 0; tap < TAPS; ++tap) {
    const int tok = t + tap - (TAPS >> 1);
    const bool valid = (unsigned)tok < (unsigned)TDIM;   // per-lane, per-tap
    const __bf16* ap = arow + (tap - (TAPS >> 1)) * CIN;
#pragma unroll 6
    for (int c0 = 0; c0 < CIN; c0 += 32) {
      const v16bf a = valid ? ldtile(ap + c0) : zero16();
      const int kk = tap * CIN + c0;
#pragma unroll
      for (int i = 0; i < 4; ++i)
        acc[i] = wmma_bf(a, ldtile(wrow[i] + kk), acc[i]);
    }
  }

  const int rowadd = (lane & 16) ? 8 : 0;
#pragma unroll
  for (int i = 0; i < 4; ++i) {
    const int n = nbase + i * 16 + lrow;
    const float bia = bias[n];
#pragma unroll
    for (int r = 0; r < 8; ++r) {
      const int m = mbase + r + rowadd;
      float v = (acc[i][r] + bia) * scale;
      if (relu) v = fmaxf(v, 0.f);
      const size_t idx = (size_t)m * NOUT + n;
      if (outF) outF[idx] = v;
      if (outB) outB[idx] = (__bf16)v;
    }
  }
}

// ---------------- attention ----------------
// One block per (b, h, 16-query tile). Full softmax over T=1024 keys.
// q pre-scaled by 1/sqrt(kc). rel_k/rel_v reduce to a +-4 diagonal band.
__global__ __launch_bounds__(256)
void attn_kernel(const __bf16* __restrict__ qbf, const __bf16* __restrict__ kbf,
                 const __bf16* __restrict__ vt,    // [B, 2, 96, T] bf16
                 const float* __restrict__ relk,   // [9, 96]
                 const float* __restrict__ relv,   // [9, 96]
                 __bf16* __restrict__ ao) {
  __shared__ float S[16][TDIM];                    // 64 KB: score / prob rows
  __shared__ __align__(16) __bf16 Pb[16][TDIM];    // 32 KB: bf16 probs
  constexpr int T = TDIM;
  const int nq  = T / 16;
  const int blk = blockIdx.x;
  const int q0  = (blk % nq) * 16;
  const int h   = (blk / nq) & 1;
  const int b   = blk / (nq * 2);
  const int lane  = threadIdx.x & 31;
  const int wave  = threadIdx.x >> 5;
  const int half8 = (lane & 16) ? 8 : 0;
  const int lrow  = lane & 15;

  // A operand: q tile (16 queries x 96 ch), 3 K-chunks, held in registers
  v16bf aq[3];
  {
    const int qrow = q0 + lrow;
    const __bf16* qp = qbf + ((size_t)(b * T + qrow)) * CDIM + h * KCD + half8;
#pragma unroll
    for (int ck = 0; ck < 3; ++ck) aq[ck] = ldtile(qp + ck * 32);
  }
  // Phase 1: scores = q . k^T -> LDS
  for (int j = wave; j < nq; j += 8) {
    v8f acc = zero8();
    const int krow = j * 16 + lrow;
    const __bf16* kp = kbf + ((size_t)(b * T + krow)) * CDIM + h * KCD + half8;
#pragma unroll
    for (int ck = 0; ck < 3; ++ck) acc = wmma_bf(aq[ck], ldtile(kp + ck * 32), acc);
    const int rowadd = (lane & 16) ? 8 : 0;
    const int col = j * 16 + lrow;
#pragma unroll
    for (int r = 0; r < 8; ++r) S[r + rowadd][col] = acc[r];
  }
  __syncthreads();
  // relative-key band bias: scores[t][s] += q[t] . rel_k[s-t+4]  for |s-t|<=4
  {
    const int tid = threadIdx.x;
    if (tid < 16 * 9) {
      const int tl = tid / 9;
      const int dt = tid % 9 - 4;
      const int tg = q0 + tl;
      const int s  = tg + dt;
      if (s >= 0 && s < T) {
        const __bf16* qp = qbf + ((size_t)(b * T + tg)) * CDIM + h * KCD;
        const float* rk = relk + (dt + 4) * KCD;
        float acc = 0.f;
#pragma unroll 8
        for (int c = 0; c < KCD; ++c) acc += (float)qp[c] * rk[c];
        S[tl][s] += acc;
      }
    }
  }
  __syncthreads();
  // softmax (2 rows per wave); also emit bf16 probs for the P.V WMMA
#pragma unroll
  for (int rr = 0; rr < 2; ++rr) {
    const int row = wave * 2 + rr;
    float mx = -3.0e38f;
    for (int s = lane; s < T; s += 32) mx = fmaxf(mx, S[row][s]);
#pragma unroll
    for (int off = 16; off > 0; off >>= 1) mx = fmaxf(mx, __shfl_xor(mx, off));
    float sum = 0.f;
    for (int s = lane; s < T; s += 32) {
      const float e = __expf(S[row][s] - mx);
      S[row][s] = e;
      sum += e;
    }
#pragma unroll
    for (int off = 16; off > 0; off >>= 1) sum += __shfl_xor(sum, off);
    const float inv = 1.f / sum;
    for (int s = lane; s < T; s += 32) {
      const float pv = S[row][s] * inv;
      S[row][s]  = pv;
      Pb[row][s] = (__bf16)pv;
    }
  }
  __syncthreads();
  // Phase 2: out = P . V (+ rel_v band). 6 waves, one 16-ch N-tile each.
  if (wave < 6) {
    const int ci = wave;
    v8f acc = zero8();
    const __bf16* prow = &Pb[lrow][0];
    const __bf16* vp =
        vt + (((size_t)(b * 2 + h)) * KCD + ci * 16 + lrow) * (size_t)T + half8;
    for (int kk = 0; kk < T; kk += 32) {
      const v16bf a = mk16(*(const v8bf*)(prow + kk + half8),
                           *(const v8bf*)(prow + kk + 16 + half8));
      acc = wmma_bf(a, ldtile(vp + kk), acc);
    }
    const int rowadd = (lane & 16) ? 8 : 0;
    const int col = ci * 16 + lrow;
#pragma unroll
    for (int r = 0; r < 8; ++r) {
      const int m  = r + rowadd;
      const int tg = q0 + m;
      float v = acc[r];
#pragma unroll
      for (int dt = -4; dt <= 4; ++dt) {
        const int s = tg + dt;
        if (s >= 0 && s < T) v += S[m][s] * relv[(dt + 4) * KCD + col];
      }
      ao[((size_t)(b * T + tg)) * CDIM + h * KCD + col] = (__bf16)v;
    }
  }
}

// ---------------- residual + LayerNorm over C (192 threads/row) ----------------
__global__ __launch_bounds__(192)
void ln_kernel(float* __restrict__ x, const float* __restrict__ y,
               const float* __restrict__ g, const float* __restrict__ be,
               __bf16* __restrict__ xbf) {
  __shared__ float red[256];
  const int row = blockIdx.x;
  const int c = threadIdx.x;
  const size_t idx = (size_t)row * CDIM + c;
  const float s = x[idx] + y[idx];
  red[c] = s;
  if (c < 64) red[CDIM + c] = 0.f;
  __syncthreads();
#pragma unroll
  for (int off = 128; off > 0; off >>= 1) {
    if (c < off) red[c] += red[c + off];
    __syncthreads();
  }
  const float mean = red[0] * (1.f / CDIM);
  __syncthreads();
  const float d = s - mean;
  red[c] = d * d;
  if (c < 64) red[CDIM + c] = 0.f;
  __syncthreads();
#pragma unroll
  for (int off = 128; off > 0; off >>= 1) {
    if (c < off) red[c] += red[c + off];
    __syncthreads();
  }
  const float var = red[0] * (1.f / CDIM);
  const float o = d * rsqrtf(var + 1e-5f) * g[c] + be[c];
  x[idx] = o;
  xbf[idx] = (__bf16)o;
}

// ---------------- small layout / conversion kernels ----------------
__global__ __launch_bounds__(256)
void tin_kernel(const float* __restrict__ xin, float* __restrict__ xbtc,
                __bf16* __restrict__ xbf, int n) {
  const int i = blockIdx.x * 256 + threadIdx.x;
  if (i >= n) return;
  const int t = i % TDIM;
  const int ch = (i / TDIM) % CDIM;
  const int b = i / (TDIM * CDIM);
  const float v = xin[i];
  const size_t o = ((size_t)b * TDIM + t) * CDIM + ch;
  xbtc[o] = v;
  xbf[o] = (__bf16)v;
}
__global__ __launch_bounds__(256)
void tout_kernel(const float* __restrict__ xbtc, float* __restrict__ out, int n) {
  const int i = blockIdx.x * 256 + threadIdx.x;
  if (i >= n) return;
  const int t = i % TDIM;
  const int ch = (i / TDIM) % CDIM;
  const int b = i / (TDIM * CDIM);
  out[i] = xbtc[((size_t)b * TDIM + t) * CDIM + ch];
}
__global__ __launch_bounds__(256)
void cvt_kernel(const float* __restrict__ src, __bf16* __restrict__ dst, int n) {
  const int i = blockIdx.x * 256 + threadIdx.x;
  if (i < n) dst[i] = (__bf16)src[i];
}
// [L,F,Cin,3] f32 -> [L,F,3*Cin] bf16  (wpack[l][f][k*Cin+c] = w[l][f][c][k])
__global__ __launch_bounds__(256)
void packconv_kernel(const float* __restrict__ w, __bf16* __restrict__ out, int F,
                     int Cin, int n) {
  const int i = blockIdx.x * 256 + threadIdx.x;
  if (i >= n) return;
  const int k = i % 3;
  const int c = (i / 3) % Cin;
  const int lf = i / (3 * Cin);  // l*F + f
  out[(size_t)lf * (3 * Cin) + k * Cin + c] = (__bf16)w[i];
}
// v [B*T, 192] bf16 -> vt [B, 2, 96, T] bf16
__global__ __launch_bounds__(256)
void vtr_kernel(const __bf16* __restrict__ v, __bf16* __restrict__ vt, int n) {
  const int i = blockIdx.x * 256 + threadIdx.x;
  if (i >= n) return;
  const int d = i % CDIM;
  const int nrow = i / CDIM;
  const int t = nrow % TDIM;
  const int b = nrow / TDIM;
  const int h = d / KCD;
  const int c = d - h * KCD;
  vt[(((size_t)b * 2 + h) * KCD + c) * TDIM + t] = v[i];
}

// ---------------- host ----------------
static inline char* carve(char*& p, size_t bytes) {
  char* r = p;
  p += (bytes + 255) & ~(size_t)255;
  return r;
}

extern "C" void kernel_launch(void* const* d_in, const int* in_sizes, int n_in,
                              void* d_out, int out_size, void* d_ws, size_t ws_size,
                              hipStream_t stream) {
  const float* x_in  = (const float*)d_in[0];
  const float* wq    = (const float*)d_in[1];
  const float* bq    = (const float*)d_in[2];
  const float* wk    = (const float*)d_in[3];
  const float* bk    = (const float*)d_in[4];
  const float* wv    = (const float*)d_in[5];
  const float* bv    = (const float*)d_in[6];
  const float* wo    = (const float*)d_in[7];
  const float* bo    = (const float*)d_in[8];
  const float* rel_k = (const float*)d_in[9];
  const float* rel_v = (const float*)d_in[10];
  const float* g1    = (const float*)d_in[11];
  const float* be1   = (const float*)d_in[12];
  const float* c1w   = (const float*)d_in[13];
  const float* c1b   = (const float*)d_in[14];
  const float* c2w   = (const float*)d_in[15];
  const float* c2b   = (const float*)d_in[16];
  const float* g2    = (const float*)d_in[17];
  const float* be2   = (const float*)d_in[18];
  float* out = (float*)d_out;

  char* p = (char*)d_ws;
  float*  xbtc = (float*)carve(p, (size_t)MROWS * CDIM * 4);
  float*  yscr = (float*)carve(p, (size_t)MROWS * CDIM * 4);
  __bf16* xbf  = (__bf16*)carve(p, (size_t)MROWS * CDIM * 2);
  __bf16* qb   = (__bf16*)carve(p, (size_t)MROWS * CDIM * 2);
  __bf16* kb   = (__bf16*)carve(p, (size_t)MROWS * CDIM * 2);
  __bf16* vb   = (__bf16*)carve(p, (size_t)MROWS * CDIM * 2);
  __bf16* vt   = (__bf16*)carve(p, (size_t)MROWS * CDIM * 2);
  __bf16* ao   = (__bf16*)carve(p, (size_t)MROWS * CDIM * 2);
  __bf16* hb   = (__bf16*)carve(p, (size_t)MROWS * FCD * 2);
  __bf16* wqb  = (__bf16*)carve(p, (size_t)LNUM * CDIM * CDIM * 2);
  __bf16* wkb  = (__bf16*)carve(p, (size_t)LNUM * CDIM * CDIM * 2);
  __bf16* wvb  = (__bf16*)carve(p, (size_t)LNUM * CDIM * CDIM * 2);
  __bf16* wob  = (__bf16*)carve(p, (size_t)LNUM * CDIM * CDIM * 2);
  __bf16* c1p  = (__bf16*)carve(p, (size_t)LNUM * FCD * (3 * CDIM) * 2);
  __bf16* c2p  = (__bf16*)carve(p, (size_t)LNUM * CDIM * (3 * FCD) * 2);

  const int nw = LNUM * CDIM * CDIM;       // 221184
  const int gw = (nw + 255) / 256;
  cvt_kernel<<<gw, 256, 0, stream>>>(wq, wqb, nw);
  cvt_kernel<<<gw, 256, 0, stream>>>(wk, wkb, nw);
  cvt_kernel<<<gw, 256, 0, stream>>>(wv, wvb, nw);
  cvt_kernel<<<gw, 256, 0, stream>>>(wo, wob, nw);
  const int nc = LNUM * FCD * CDIM * 3;    // 2654208 (same for both convs)
  packconv_kernel<<<(nc + 255) / 256, 256, 0, stream>>>(c1w, c1p, FCD, CDIM, nc);
  packconv_kernel<<<(nc + 255) / 256, 256, 0, stream>>>(c2w, c2p, CDIM, FCD, nc);

  const int nx = BDIM * CDIM * TDIM;       // 1572864
  tin_kernel<<<(nx + 255) / 256, 256, 0, stream>>>(x_in, xbtc, xbf, nx);

  const float sc = 1.0f / sqrtf((float)KCD);
  const dim3 blk(256);
  const dim3 gP(MROWS / 128, CDIM / 64);   // 64 x 3
  const dim3 gF(MROWS / 128, FCD / 64);    // 64 x 12

  for (int l = 0; l < LNUM; ++l) {
    const size_t wcc = (size_t)l * CDIM * CDIM;
    // Q / K / V projections (q pre-scaled by 1/sqrt(kc))
    gemm_tpl<CDIM, 1, CDIM><<<gP, blk, 0, stream>>>(xbf, wqb + wcc, bq + l * CDIM,
                                                    nullptr, qb, sc, 0);
    gemm_tpl<CDIM, 1, CDIM><<<gP, blk, 0, stream>>>(xbf, wkb + wcc, bk + l * CDIM,
                                                    nullptr, kb, 1.f, 0);
    gemm_tpl<CDIM, 1, CDIM><<<gP, blk, 0, stream>>>(xbf, wvb + wcc, bv + l * CDIM,
                                                    nullptr, vb, 1.f, 0);
    vtr_kernel<<<(nx + 255) / 256, 256, 0, stream>>>(vb, vt, nx);
    // attention (rel_k / rel_v are [L,1,9,96])
    attn_kernel<<<BDIM * 2 * (TDIM / 16), blk, 0, stream>>>(
        qb, kb, vt, rel_k + (size_t)l * 9 * KCD, rel_v + (size_t)l * 9 * KCD, ao);
    // output projection -> f32 y, then residual + LN1
    gemm_tpl<CDIM, 1, CDIM><<<gP, blk, 0, stream>>>(ao, wob + wcc, bo + l * CDIM,
                                                    yscr, nullptr, 1.f, 0);
    ln_kernel<<<MROWS, 192, 0, stream>>>(xbtc, yscr, g1 + l * CDIM, be1 + l * CDIM,
                                         xbf);
    // FFN: conv1 (C->Fc, K=3, relu) then conv2 (Fc->C, K=3)
    gemm_tpl<CDIM, 3, FCD><<<gF, blk, 0, stream>>>(
        xbf, c1p + (size_t)l * FCD * (3 * CDIM), c1b + l * FCD, nullptr, hb, 1.f, 1);
    gemm_tpl<FCD, 3, CDIM><<<gP, blk, 0, stream>>>(
        hb, c2p + (size_t)l * CDIM * (3 * FCD), c2b + l * CDIM, yscr, nullptr, 1.f,
        0);
    ln_kernel<<<MROWS, 192, 0, stream>>>(xbtc, yscr, g2 + l * CDIM, be2 + l * CDIM,
                                         xbf);
  }
  tout_kernel<<<(nx + 255) / 256, 256, 0, stream>>>(xbtc, out, nx);
}
